// Attention_17729624998444
// MI455X (gfx1250) — matrix operand order repaired and checked by reading, not compiled, not run on hardware
//
#include <hip/hip_runtime.h>

typedef _Float16 half_t;
typedef __attribute__((ext_vector_type(16))) _Float16 v16h;
typedef __attribute__((ext_vector_type(8)))  _Float16 v8h;
typedef __attribute__((ext_vector_type(4)))  _Float16 v4h;
typedef __attribute__((ext_vector_type(8)))  float    v8f;
typedef __attribute__((ext_vector_type(4)))  int      v4i;

// ---- CDNA5 async global->LDS copy (ASYNCcnt path), with safe fallback ----
#if defined(__has_builtin)
#if __has_builtin(__builtin_amdgcn_global_load_async_to_lds_b128)
#define HAVE_ASYNC_LDS 1
#endif
#if __has_builtin(__builtin_amdgcn_s_wait_asynccnt)
#define HAVE_WAIT_ASYNC_BUILTIN 1
#endif
#endif
#ifndef HAVE_ASYNC_LDS
#define HAVE_ASYNC_LDS 0
#endif
#ifndef HAVE_WAIT_ASYNC_BUILTIN
#define HAVE_WAIT_ASYNC_BUILTIN 0
#endif

#if HAVE_ASYNC_LDS
typedef __attribute__((address_space(1))) v4i as1_v4i;
typedef __attribute__((address_space(3))) v4i as3_v4i;

__device__ inline void async_copy16(half_t* lds_dst, const void* gsrc) {
  // generic->AS1/AS3: AMDGPU generic pointers carry the LDS offset in the
  // low 32 bits (ISA 10.2 aperture mapping); global is bit-identical.
  __builtin_amdgcn_global_load_async_to_lds_b128(
      (as1_v4i*)(unsigned long long)gsrc,
      (as3_v4i*)(unsigned)(unsigned long long)lds_dst,
      0, 0);
}
__device__ inline void async_wait0() {
#if HAVE_WAIT_ASYNC_BUILTIN
  __builtin_amdgcn_s_wait_asynccnt(0);
#else
  asm volatile("s_wait_asynccnt 0" ::: "memory");
#endif
}
#endif

__device__ inline v8f zero_v8f() {
  v8f z;
#pragma unroll
  for (int j = 0; j < 8; ++j) z[j] = 0.0f;
  return z;
}

__device__ inline v16h cat16(v8h lo, v8h hi) {
  return __builtin_shufflevector(lo, hi, 0, 1, 2, 3, 4, 5, 6, 7,
                                 8, 9, 10, 11, 12, 13, 14, 15);
}

__device__ inline v8f wmma_f16(v16h a, v16h b, v8f c) {
  // D = A*B + C, fp32 accum; emits v_wmma_f32_16x16x32_f16
  return __builtin_amdgcn_wmma_f32_16x16x32_f16(
      /*neg_a=*/false, a, /*neg_b=*/false, b,
      /*c_mod=*/(short)0, c, /*reuse_a=*/false, /*reuse_b=*/false);
}

// lane exchange via ds_swizzle_b32 (immediate xor pattern, no index VGPR):
// group-of-32 encoding: offset[14:10]=xor, offset[9:5]=or, offset[4:0]=and
template <int M>
__device__ inline float swz_xor(float v) {
  return __int_as_float(
      __builtin_amdgcn_ds_swizzle(__float_as_int(v), (M << 10) | 0x1f));
}
// reductions over the 16 lanes of a lane-half (xor masks stay within half)
__device__ inline float redmax16(float v) {
  v = fmaxf(v, swz_xor<1>(v));
  v = fmaxf(v, swz_xor<2>(v));
  v = fmaxf(v, swz_xor<4>(v));
  v = fmaxf(v, swz_xor<8>(v));
  return v;
}
__device__ inline float redsum16(float v) {
  v += swz_xor<1>(v);
  v += swz_xor<2>(v);
  v += swz_xor<4>(v);
  v += swz_xor<8>(v);
  return v;
}

// ---------------------------------------------------------------------------
// Tiled WMMA GEMM: C[M,N] = A[M,K] (f32 or f16) * B[K,N] (f32)
// Block: 256 thr (8 waves), tile 128x128x32. Wave grid 2(M)x4(N),
// each wave computes 64x32 = 4x2 WMMA tiles.
// A tile staging: f16 source -> GLOBAL_LOAD_ASYNC_TO_LDS_B128 (ASYNCcnt),
//                 f32 source -> float4 loads + packed f16 converts.
// EPI 0: scatter qkv (h,d,3)-interleaved cols into Q,K [bh,s,d] and Vt [bh,d,s]
// EPI 1: plain fp32 row-major store
// ---------------------------------------------------------------------------
template <typename AT, int EPI>
__global__ __launch_bounds__(256) void gemm_wmma(
    const AT* __restrict__ A, const float* __restrict__ B,
    int K, int N,
    half_t* __restrict__ Qb, half_t* __restrict__ Kb, half_t* __restrict__ Vt,
    float* __restrict__ Out) {
  constexpr int BM = 128, BN = 128, BK = 32, LDK = 48;  // 96B padded stride
  __shared__ half_t sA[BM * LDK] __attribute__((aligned(32)));
  __shared__ half_t sB[BN * LDK] __attribute__((aligned(32)));  // sB[n][k]

  const int tid = threadIdx.x;
  const int lane = tid & 31;
  const int wave = tid >> 5;
  const int l16 = lane & 15;
  const int lh = lane >> 4;
  const int wm = wave & 1;
  const int wn = wave >> 1;
  const int m0 = blockIdx.y * BM;
  const int n0 = blockIdx.x * BN;

  v8f acc[4][2];
#pragma unroll
  for (int mt = 0; mt < 4; ++mt)
#pragma unroll
    for (int nt = 0; nt < 2; ++nt) acc[mt][nt] = zero_v8f();

  for (int k0 = 0; k0 < K; k0 += BK) {
    // ---- stage A tile (BM x BK halves) ----
    if constexpr (sizeof(AT) == 2) {
#pragma unroll
      for (int j = 0; j < 2; ++j) {  // 512 x 16B chunks, 2 per thread
        int p = tid + 256 * j;
        int r = p >> 2, c = (p & 3) * 8;
        const AT* src = A + (size_t)(m0 + r) * K + (k0 + c);
        half_t* dst = &sA[r * LDK + c];
#if HAVE_ASYNC_LDS
        async_copy16(dst, (const void*)src);
#else
        *(v8h*)dst = *(const v8h*)src;
#endif
      }
    } else {
#pragma unroll
      for (int j = 0; j < 4; ++j) {  // 1024 float4, 4 per thread
        int p = tid + 256 * j;
        int r = p >> 3, c = (p & 7) * 4;
        float4 f = *(const float4*)(A + (size_t)(m0 + r) * K + (k0 + c));
        v4h h;
        h[0] = (half_t)f.x; h[1] = (half_t)f.y;
        h[2] = (half_t)f.z; h[3] = (half_t)f.w;
        *(v4h*)&sA[r * LDK + c] = h;
      }
    }
    // ---- stage B tile (BK x BN) transposed into sB[n][k], f32 -> f16 ----
#pragma unroll
    for (int j = 0; j < 4; ++j) {  // 1024 float4, 4 per thread
      int p = tid + 256 * j;
      int k = p >> 5, n = (p & 31) * 4;
      float4 f = *(const float4*)(B + (size_t)(k0 + k) * N + (n0 + n));
      sB[(n + 0) * LDK + k] = (half_t)f.x;
      sB[(n + 1) * LDK + k] = (half_t)f.y;
      sB[(n + 2) * LDK + k] = (half_t)f.z;
      sB[(n + 3) * LDK + k] = (half_t)f.w;
    }
    if (k0 + BK < K) {  // warm L2 for the next k-step (global_prefetch_b8)
      __builtin_prefetch(&A[(size_t)(m0 + (tid >> 1)) * K + (k0 + BK)], 0, 0);
      __builtin_prefetch(&B[(size_t)(k0 + BK + (tid >> 4)) * N + n0 + (tid & 15) * 8], 0, 0);
    }
#if HAVE_ASYNC_LDS
    if constexpr (sizeof(AT) == 2) async_wait0();  // own async writes done
#endif
    __syncthreads();

    // A fragment: lane row = M, K contiguous lh*16..+15 (matches B order)
    v16h af[4], bf[2];
#pragma unroll
    for (int mt = 0; mt < 4; ++mt) {
      const half_t* p = sA + (wm * 64 + mt * 16 + l16) * LDK + lh * 16;
      af[mt] = cat16(*(const v8h*)p, *(const v8h*)(p + 8));
    }
    // B fragment: lane = N column, K contiguous lh*16..+15 (same order as A)
#pragma unroll
    for (int nt = 0; nt < 2; ++nt) {
      const half_t* p = sB + (wn * 32 + nt * 16 + l16) * LDK + lh * 16;
      bf[nt] = *(const v16h*)p;
    }
#pragma unroll
    for (int mt = 0; mt < 4; ++mt)
#pragma unroll
      for (int nt = 0; nt < 2; ++nt)
        acc[mt][nt] = wmma_f16(af[mt], bf[nt], acc[mt][nt]);
    __syncthreads();
  }

  // Epilogue. C layout: VGPR i holds M=i (lanes 0-15) / M=i+8 (lanes 16-31)
#pragma unroll
  for (int mt = 0; mt < 4; ++mt) {
#pragma unroll
    for (int nt = 0; nt < 2; ++nt) {
#pragma unroll
      for (int i = 0; i < 8; ++i) {
        int grow = m0 + wm * 64 + mt * 16 + i + lh * 8;
        int gcol = n0 + wn * 32 + nt * 16 + l16;
        float v = acc[mt][nt][i];
        if constexpr (EPI == 0) {
          // column c -> reshape(h, d, 3): which = c%3, d = (c/3)%64, h = c/192
          int which = gcol % 3;
          int dd = (gcol / 3) & 63;
          int hh = gcol / 192;
          int b = grow >> 11;
          int s = grow & 2047;
          int bh = b * 16 + hh;
          half_t hv = (half_t)v;
          if (which == 0)
            Qb[((size_t)bh * 2048 + s) * 64 + dd] = hv;
          else if (which == 1)
            Kb[((size_t)bh * 2048 + s) * 64 + dd] = hv;
          else
            Vt[((size_t)bh * 64 + dd) * 2048 + s] = hv;  // V transposed
        } else {
          Out[(size_t)grow * N + gcol] = v;
        }
      }
    }
  }
}

// ---------------------------------------------------------------------------
// Flash attention, one (b,h) per blockIdx.y, 64 queries per block (4 waves x
// 16 queries). Streams 32-key blocks: S = Q Kt (2x2 WMMA), online softmax
// with ds_swizzle row reductions, P -> LDS -> A-fragment, O += P V (4 WMMA).
// Writes attn as f16 [b*s, h*d] row-major for the final GEMM.
// ---------------------------------------------------------------------------
__global__ __launch_bounds__(128) void attn_kernel(
    const half_t* __restrict__ Qb, const half_t* __restrict__ Kb,
    const half_t* __restrict__ Vt, half_t* __restrict__ Ab) {
  __shared__ half_t sP[4][16 * 32] __attribute__((aligned(32)));
  const int tid = threadIdx.x;
  const int lane = tid & 31;
  const int wave = tid >> 5;
  const int l16 = lane & 15;
  const int lh = lane >> 4;
  const int bh = blockIdx.y;
  const int b = bh >> 4, h = bh & 15;
  const int qbase = blockIdx.x * 64 + wave * 16;

  const half_t* Qp = Qb + (size_t)bh * 2048 * 64;
  const half_t* Kp = Kb + (size_t)bh * 2048 * 64;
  const half_t* Vtp = Vt + (size_t)bh * 64 * 2048;
  half_t* sPw = sP[wave];

  // Q fragments (A operand), one per 32-wide d-chunk; reused all iterations
  v16h qf[2];
#pragma unroll
  for (int dc = 0; dc < 2; ++dc) {
    const half_t* p = Qp + (size_t)(qbase + l16) * 64 + dc * 32 + lh * 16;
    qf[dc] = cat16(*(const v8h*)p, *(const v8h*)(p + 8));
  }

  float mrow[16], lrow[16];  // per-row running max / sum, replicated per lane
#pragma unroll
  for (int i = 0; i < 16; ++i) { mrow[i] = -1e30f; lrow[i] = 0.0f; }
  v8f O[4];
#pragma unroll
  for (int t = 0; t < 4; ++t) O[t] = zero_v8f();

  for (int kb = 0; kb < 2048; kb += 32) {
    // ---- S = Q K^T for 32 keys (two 16-key n-tiles) ----
    v8f s0 = zero_v8f(), s1 = zero_v8f();
#pragma unroll
    for (int dc = 0; dc < 2; ++dc) {
      // B operand of QK^T: lane = key column, d contiguous -> K row-major read
      const half_t* pk0 = Kp + (size_t)(kb + l16) * 64 + dc * 32 + lh * 16;
      const half_t* pk1 = Kp + (size_t)(kb + 16 + l16) * 64 + dc * 32 + lh * 16;
      v16h kf0 = *(const v16h*)pk0;
      v16h kf1 = *(const v16h*)pk1;
      s0 = wmma_f16(qf[dc], kf0, s0);
      s1 = wmma_f16(qf[dc], kf1, s1);
    }

    // ---- online softmax (rows split: VGPR i -> q=i / q=i+8 by lane half) ----
    float oal[8];
#pragma unroll
    for (int i = 0; i < 8; ++i) {
      s0[i] *= 0.125f;  // 1/sqrt(64)
      s1[i] *= 0.125f;
      float rm = redmax16(fmaxf(s0[i], s1[i]));
      float rmo = swz_xor<16>(rm);
      float mk_lo = (lane < 16) ? rm : rmo;   // block max of row i
      float mk_hi = (lane < 16) ? rmo : rm;   // block max of row i+8
      float mn_lo = fmaxf(mrow[i], mk_lo);
      float mn_hi = fmaxf(mrow[i + 8], mk_hi);
      float al_lo = __expf(mrow[i] - mn_lo);
      float al_hi = __expf(mrow[i + 8] - mn_hi);
      mrow[i] = mn_lo;
      mrow[i + 8] = mn_hi;
      float mr = (lane < 16) ? mn_lo : mn_hi;
      float p0 = __expf(s0[i] - mr);
      float p1 = __expf(s1[i] - mr);
      s0[i] = p0;
      s1[i] = p1;
      float rs = redsum16(p0 + p1);
      float rso = swz_xor<16>(rs);
      lrow[i] = lrow[i] * al_lo + ((lane < 16) ? rs : rso);
      lrow[i + 8] = lrow[i + 8] * al_hi + ((lane < 16) ? rso : rs);
      oal[i] = (lane < 16) ? al_lo : al_hi;  // rescale factor for O row
    }
#pragma unroll
    for (int t = 0; t < 4; ++t)
#pragma unroll
      for (int i = 0; i < 8; ++i) O[t][i] *= oal[i];

    // ---- P: C layout -> LDS -> A-fragment (wave-private; LDS is in-order) --
#pragma unroll
    for (int i = 0; i < 8; ++i) {
      int r = i + lh * 8;
      sPw[r * 32 + l16] = (half_t)s0[i];
      sPw[r * 32 + 16 + l16] = (half_t)s1[i];
    }
    const half_t* pp = sPw + l16 * 32 + lh * 16;
    v16h pf = cat16(*(const v8h*)pp, *(const v8h*)(pp + 8));

    // ---- O += P V : B operand lane = d column, k contiguous via Vt --------
#pragma unroll
    for (int t = 0; t < 4; ++t) {
      const half_t* pv = Vtp + (size_t)(t * 16 + l16) * 2048 + kb + lh * 16;
      v16h vb = *(const v16h*)pv;
      O[t] = wmma_f16(pf, vb, O[t]);
    }
  }

  // ---- normalize and write attn[b*s, h*64+d] as f16 ----
#pragma unroll
  for (int t = 0; t < 4; ++t)
#pragma unroll
    for (int i = 0; i < 8; ++i) {
      int q = qbase + i + lh * 8;
      float linv = 1.0f / ((lane < 16) ? lrow[i] : lrow[i + 8]);
      int col = h * 64 + t * 16 + l16;
      Ab[((size_t)b * 2048 + q) * 1024 + col] = (half_t)(O[t][i] * linv);
    }
}

// ---------------------------------------------------------------------------
extern "C" void kernel_launch(void* const* d_in, const int* in_sizes, int n_in,
                              void* d_out, int out_size, void* d_ws,
                              size_t ws_size, hipStream_t stream) {
  const float* X = (const float*)d_in[0];      // [2,2048,1024]
  const float* Wqkv = (const float*)d_in[1];   // [1024,3072]
  const float* Wout = (const float*)d_in[2];   // [1024,1024]
  float* Out = (float*)d_out;                  // [2,2048,1024]

  // workspace: Q,K [bh,s,d], Vt [bh,d,s], attn [b*s, h*d]; 4 x 8MB f16
  const size_t SZ = (size_t)32 * 2048 * 64;  // 4,194,304 halves each
  half_t* Qb = (half_t*)d_ws;
  half_t* Kb = Qb + SZ;
  half_t* Vt = Kb + SZ;
  half_t* Ab = Vt + SZ;

  // 1) QKV projection + de-interleave scatter
  gemm_wmma<float, 0><<<dim3(3072 / 128, 4096 / 128), 256, 0, stream>>>(
      X, Wqkv, /*K=*/1024, /*N=*/3072, Qb, Kb, Vt, nullptr);

  // 2) flash attention: 32 q-tiles x 32 (b,h)
  attn_kernel<<<dim3(32, 32), 128, 0, stream>>>(Qb, Kb, Vt, Ab);

  // 3) output projection -> fp32 (A tiles via async global->LDS DMA)
  gemm_wmma<half_t, 1><<<dim3(1024 / 128, 4096 / 128), 256, 0, stream>>>(
      Ab, Wout, /*K=*/1024, /*N=*/1024, nullptr, nullptr, nullptr, Out);
}
